// LAMA_18794776887708
// MI455X (gfx1250) — compile-verified, matrix operand order checked by reading, not run on hardware
//
#include <hip/hip_runtime.h>
#include <hip/hip_bf16.h>
#include <math.h>

// ---------------- CDNA5 WMMA bf16 helpers (wave32) ----------------
typedef __bf16 v16bf __attribute__((ext_vector_type(16)));
typedef __bf16 v8bf  __attribute__((ext_vector_type(8)));
typedef __bf16 v4bf  __attribute__((ext_vector_type(4)));
typedef float  v8f   __attribute__((ext_vector_type(8)));

__device__ __forceinline__ v8f wmma_bf16(v16bf a, v16bf b, v8f c) {
  // D = A(16x32 bf16) * B(32x16 bf16) + C(16x16 f32)
  return __builtin_amdgcn_wmma_f32_16x16x32_bf16(
      false, a, false, b, (short)0, c, false, false);
}

union FragU { v16bf v; v8bf h[2]; };

// Fragment gather for the ISA A-layout (7.12.2) from a ROW-MAJOR bf16 tile:
// lane L(<16): row=L, K = {0..7} in VGPR0-3 and {16..23} in VGPR4-7;
// lane L+16:   row=L, K = {8..15} / {24..31}.
// => per lane: two contiguous 16B loads at row*ld + half*8 (+16).
// Works for A tiles and for B tiles stored TRANSPOSED ([n][k]).
// Requires ld % 8 == 0 (16B alignment).
__device__ __forceinline__ v16bf load_frag_bf16(const __bf16* __restrict__ p,
                                                int ld, int lane) {
  const __bf16* row = p + (lane & 15) * ld + ((lane >> 4) << 3);
  FragU f;
  f.h[0] = *(const v8bf*)(row);
  f.h[1] = *(const v8bf*)(row + 16);
  return f.v;
}

// Same fragment pattern, loading fp32 from global with explicit float4
// (addresses are 16B aligned by construction) and packed conversion.
__device__ __forceinline__ v16bf load_frag_f32(const float* __restrict__ p,
                                               long long ld, int lane) {
  const float* row = p + (long long)(lane & 15) * ld + ((lane >> 4) << 3);
  float4 a0 = *(const float4*)(row);
  float4 a1 = *(const float4*)(row + 4);
  float4 b0 = *(const float4*)(row + 16);
  float4 b1 = *(const float4*)(row + 20);
  v16bf r;
  r[0]  = (__bf16)a0.x; r[1]  = (__bf16)a0.y; r[2]  = (__bf16)a0.z; r[3]  = (__bf16)a0.w;
  r[4]  = (__bf16)a1.x; r[5]  = (__bf16)a1.y; r[6]  = (__bf16)a1.z; r[7]  = (__bf16)a1.w;
  r[8]  = (__bf16)b0.x; r[9]  = (__bf16)b0.y; r[10] = (__bf16)b0.z; r[11] = (__bf16)b0.w;
  r[12] = (__bf16)b1.x; r[13] = (__bf16)b1.y; r[14] = (__bf16)b1.z; r[15] = (__bf16)b1.w;
  return r;
}

// ---------------- problem constants ----------------
constexpr int BDIM = 16, HDIM = 64, WDIM = 64, CDIM = 384;
constexpr int NHEAD = 8, DHEAD = 48, NSEQ = HDIM * WDIM;           // 4096
constexpr long long TTOK = (long long)BDIM * NSEQ;                 // 65536
constexpr long long TC = TTOK * CDIM;

enum { ACT_NONE = 0, ACT_SILU = 1, ACT_GELU = 2, ACT_ELU1 = 3 };

__device__ __forceinline__ float apply_act(float v, int ACT) {
  if (ACT == ACT_SILU) return v / (1.0f + __expf(-v));
  if (ACT == ACT_GELU) return 0.5f * v * (1.0f + erff(v * 0.70710678118654752f));
  if (ACT == ACT_ELU1) return (v > 0.0f) ? v + 1.0f : __expf(v);  // elu(x)+1
  return v;
}

// ---------------- tiled bf16 WMMA GEMM: Y = act(X@W + bias) [+ resid] ----
// X: MxK fp32, W: KxN fp32 (row-major). BM=128, BN=64, BK=32.
// 256 threads = 8 waves (4 along M x 2 along N), 32x32 per wave (2x2 WMMA).
// LDS double-buffered; B tile stored transposed so frags are 2x ds_load_b128.
template <int ACT, bool RES>
__global__ void __launch_bounds__(256) gemm_kernel(
    const float* __restrict__ X, const float* __restrict__ W,
    const float* __restrict__ bias, const float* __restrict__ resid,
    float* __restrict__ Y, int M, int K, int N) {
  constexpr int BM = 128, BN = 64, BK = 32;
  constexpr int LDA = BK + 8;  // 40 bf16 (80B rows, 16B aligned)
  constexpr int LDB = BK + 8;  // Bt is [n][k]
  __shared__ __bf16 As[2][BM * LDA];
  __shared__ __bf16 Bt[2][BN * LDB];

  const int tid = threadIdx.x, lane = tid & 31, wave = tid >> 5;
  const int wm = wave >> 1;  // 0..3
  const int wn = wave & 1;   // 0..1
  const long long bm = (long long)blockIdx.y * BM;
  const int bn = blockIdx.x * BN;

  // --- precomputed staging slots (hoisted out of the K loop) ---
  // A: 128x32 fp32 -> 1024 float4 slots, 4 per thread (contiguous b64 stores)
  const float* aptr[4];
  int aoff[4];
#pragma unroll
  for (int it = 0; it < 4; ++it) {
    int slot = tid + it * 256;
    int r = slot >> 3, cq = (slot & 7) * 4;
    aptr[it] = X + (bm + r) * (long long)K + cq;
    aoff[it] = r * LDA + cq;
  }
  // B: 32x64 fp32 -> 512 float4 slots, 2 per thread; transpose on store
  const float* bptr[2];
  int boff[2];
#pragma unroll
  for (int it = 0; it < 2; ++it) {
    int slot = tid + it * 256;
    int kk = slot >> 4, nq = (slot & 15) * 4;
    bptr[it] = W + (long long)kk * N + bn + nq;
    boff[it] = nq * LDB + kk;  // dest row = n, col = k
  }

  auto stage = [&](int k0, int buf) {
#pragma unroll
    for (int it = 0; it < 4; ++it) {
      float4 v = *(const float4*)(aptr[it] + k0);
      v4bf t;
      t[0] = (__bf16)v.x; t[1] = (__bf16)v.y; t[2] = (__bf16)v.z; t[3] = (__bf16)v.w;
      *(v4bf*)(As[buf] + aoff[it]) = t;  // one ds_store_b64
    }
#pragma unroll
    for (int it = 0; it < 2; ++it) {
      float4 v = *(const float4*)(bptr[it] + (long long)k0 * N);
      __bf16* d = Bt[buf] + boff[it];
      d[0 * LDB] = (__bf16)v.x; d[1 * LDB] = (__bf16)v.y;
      d[2 * LDB] = (__bf16)v.z; d[3 * LDB] = (__bf16)v.w;
    }
  };

  v8f c00 = {}, c01 = {}, c10 = {}, c11 = {};
  stage(0, 0);
  __syncthreads();
  int buf = 0;
  for (int k0 = 0; k0 < K; k0 += BK) {
    if (k0 + BK < K) stage(k0 + BK, buf ^ 1);  // prefetch next tile
    v16bf a0 = load_frag_bf16(As[buf] + (wm * 32) * LDA, LDA, lane);
    v16bf a1 = load_frag_bf16(As[buf] + (wm * 32 + 16) * LDA, LDA, lane);
    v16bf b0 = load_frag_bf16(Bt[buf] + (wn * 32) * LDB, LDB, lane);
    v16bf b1 = load_frag_bf16(Bt[buf] + (wn * 32 + 16) * LDB, LDB, lane);
    c00 = wmma_bf16(a0, b0, c00);
    c01 = wmma_bf16(a0, b1, c01);
    c10 = wmma_bf16(a1, b0, c10);
    c11 = wmma_bf16(a1, b1, c11);
    __syncthreads();
    buf ^= 1;
  }

  const int half = lane >> 4, nc = lane & 15;
  auto store = [&](v8f c, int mb, int nb) {
    int col = bn + wn * 32 + nb + nc;
    float bv = bias[col];
#pragma unroll
    for (int r = 0; r < 8; ++r) {
      long long row = bm + wm * 32 + mb + half * 8 + r;
      float v = apply_act(c[r] + bv, ACT);
      if (RES) v += resid[row * (long long)N + col];
      Y[row * (long long)N + col] = v;
    }
  };
  store(c00, 0, 0);  store(c01, 0, 16);
  store(c10, 16, 0); store(c11, 16, 16);
}

// ---------------- depthwise 3x3 conv on (B,H,W,C), 4 ch per thread -------
// MODE 0: out = in + conv(in)+b            (CPE residual)
// MODE 1: out = silu(conv(in)+b)           (dwc branch)
// MODE 2: out = (extra + conv(in)+b)*gate  (lepe + attn, gated by act_res)
template <int MODE>
__global__ void __launch_bounds__(256) dwconv_kernel(
    const float* __restrict__ in, const float* __restrict__ w,
    const float* __restrict__ bias, const float* __restrict__ extra,
    const float* __restrict__ gate, float* __restrict__ out) {
  constexpr int C4 = CDIM / 4;
  long long idx = (long long)blockIdx.x * 256 + threadIdx.x;
  int c4 = (int)(idx % C4);
  long long p = idx / C4;
  int xw = (int)(p % WDIM);
  long long p2 = p / WDIM;
  int yh = (int)(p2 % HDIM);
  int b = (int)(p2 / HDIM);
  int c = c4 * 4;

  float acc0 = bias[c], acc1 = bias[c + 1], acc2 = bias[c + 2], acc3 = bias[c + 3];
#pragma unroll
  for (int dy = 0; dy < 3; ++dy) {
    int iy = yh + dy - 1;
    if (iy < 0 || iy >= HDIM) continue;
#pragma unroll
    for (int dx = 0; dx < 3; ++dx) {
      int ix = xw + dx - 1;
      if (ix < 0 || ix >= WDIM) continue;
      const float* src = in + (((long long)(b * HDIM + iy) * WDIM) + ix) * CDIM + c;
      float4 v = *(const float4*)src;
      int ko = dy * 3 + dx;
      acc0 += v.x * w[(c + 0) * 9 + ko];
      acc1 += v.y * w[(c + 1) * 9 + ko];
      acc2 += v.z * w[(c + 2) * 9 + ko];
      acc3 += v.w * w[(c + 3) * 9 + ko];
    }
  }
  long long o = (((long long)(b * HDIM + yh) * WDIM) + xw) * CDIM + c;
  float4 r;
  if (MODE == 0) {
    float4 cv = *(const float4*)(in + o);
    r.x = cv.x + acc0; r.y = cv.y + acc1; r.z = cv.z + acc2; r.w = cv.w + acc3;
  } else if (MODE == 1) {
    r.x = apply_act(acc0, ACT_SILU); r.y = apply_act(acc1, ACT_SILU);
    r.z = apply_act(acc2, ACT_SILU); r.w = apply_act(acc3, ACT_SILU);
  } else {
    float4 e = *(const float4*)(extra + o);
    float4 g = *(const float4*)(gate + o);
    r.x = (e.x + acc0) * g.x; r.y = (e.y + acc1) * g.y;
    r.z = (e.z + acc2) * g.z; r.w = (e.w + acc3) * g.w;
  }
  *(float4*)(out + o) = r;
}

// ---------------- LayerNorm over C=384, one wave per token ---------------
__global__ void __launch_bounds__(256) ln_kernel(
    const float* __restrict__ x, const float* __restrict__ g,
    const float* __restrict__ b, float* __restrict__ y) {
  const int lane = threadIdx.x & 31, wave = threadIdx.x >> 5;
  long long t = (long long)blockIdx.x * 8 + wave;
  const float* row = x + t * CDIM;
  float4 v[3];
  float s = 0.0f;
#pragma unroll
  for (int i = 0; i < 3; ++i) {
    v[i] = *(const float4*)(row + lane * 4 + i * 128);
    s += v[i].x + v[i].y + v[i].z + v[i].w;
  }
#pragma unroll
  for (int off = 16; off > 0; off >>= 1) s += __shfl_xor(s, off, 32);
  float mean = s * (1.0f / CDIM);
  float vs = 0.0f;
#pragma unroll
  for (int i = 0; i < 3; ++i) {
    float d0 = v[i].x - mean, d1 = v[i].y - mean, d2 = v[i].z - mean, d3 = v[i].w - mean;
    vs += d0 * d0 + d1 * d1 + d2 * d2 + d3 * d3;
  }
#pragma unroll
  for (int off = 16; off > 0; off >>= 1) vs += __shfl_xor(vs, off, 32);
  float inv = rsqrtf(vs * (1.0f / CDIM) + 1e-5f);
#pragma unroll
  for (int i = 0; i < 3; ++i) {
    int c = lane * 4 + i * 128;
    float4 gg = *(const float4*)(g + c);
    float4 bb = *(const float4*)(b + c);
    float4 r;
    r.x = (v[i].x - mean) * inv * gg.x + bb.x;
    r.y = (v[i].y - mean) * inv * gg.y + bb.y;
    r.z = (v[i].z - mean) * inv * gg.z + bb.z;
    r.w = (v[i].w - mean) * inv * gg.w + bb.w;
    *(float4*)(y + t * CDIM + c) = r;
  }
}

// ---------------- kv = k^T v / N per (b,h), ones column -> kmean ----------
// One block per (b,h). 384 threads = 12 waves (3 M-tiles x 4 N-tiles).
// K-step = 64 tokens (2 WMMAs per wave per iter), ping-pong LDS, one
// barrier per iteration; staging is two float4 per tensor per thread.
// Output kvt[bh] is TRANSPOSED bf16 [dv(64)][dk(64)]:
//   rows dv<48: kv/N over dk<48; row 48 = kmean; rows 49..63 = 0 (B was 0);
//   cols dk 48..63 zero-filled (K padding for the out kernel).
__global__ void __launch_bounds__(384) kv_kernel(
    const float* __restrict__ qk, const float* __restrict__ hs,
    __bf16* __restrict__ kvt) {
  constexpr int LDT = 72;  // 64 tokens + 8 pad (144B rows, 16B aligned)
  const int bh = blockIdx.x, b = bh >> 3, h = bh & 7;
  const int tid = threadIdx.x, lane = tid & 31, wave = tid >> 5;
  const int wm = wave >> 2, wn = wave & 3;
  __shared__ __bf16 Ks[2][48 * LDT];  // [dk][tok]
  __shared__ __bf16 Vt[2][64 * LDT];  // [dv][tok]; rows 48..63 static
  const long long tokBase = (long long)b * NSEQ;

  // static rows in both buffers: dv=48 -> ones (kmean), dv 49..63 -> zeros
  for (int i = tid; i < 16 * 64; i += 384) {
    int dv = 48 + (i >> 6), tn = i & 63;
    __bf16 val = (dv == 48) ? (__bf16)1.0f : (__bf16)0.0f;
    Vt[0][dv * LDT + tn] = val;
    Vt[1][dv * LDT + tn] = val;
  }

  // staging slots: 64 tok x 48 ch = 768 float4 per tensor, 2 per thread
  const float* kgp[2]; const float* vgp[2];
  int lso[2];
#pragma unroll
  for (int s = 0; s < 2; ++s) {
    int slot = tid + s * 384;
    int tn = slot / 12, chq = (slot - tn * 12) * 4;
    kgp[s] = qk + (tokBase + tn) * (2 * CDIM) + CDIM + h * DHEAD + chq;
    vgp[s] = hs + (tokBase + tn) * CDIM + h * DHEAD + chq;
    lso[s] = chq * LDT + tn;
  }

  auto stage = [&](int n0, int buf) {
#pragma unroll
    for (int s = 0; s < 2; ++s) {
      float4 kv4 = *(const float4*)(kgp[s] + (long long)n0 * (2 * CDIM));
      __bf16* d = Ks[buf] + lso[s];
      d[0 * LDT] = (__bf16)kv4.x; d[1 * LDT] = (__bf16)kv4.y;
      d[2 * LDT] = (__bf16)kv4.z; d[3 * LDT] = (__bf16)kv4.w;
    }
#pragma unroll
    for (int s = 0; s < 2; ++s) {
      float4 vv4 = *(const float4*)(vgp[s] + (long long)n0 * CDIM);
      __bf16* d = Vt[buf] + lso[s];
      d[0 * LDT] = (__bf16)vv4.x; d[1 * LDT] = (__bf16)vv4.y;
      d[2 * LDT] = (__bf16)vv4.z; d[3 * LDT] = (__bf16)vv4.w;
    }
  };

  v8f acc = {};
  stage(0, 0);
  __syncthreads();
  int buf = 0;
  for (int n0 = 0; n0 < NSEQ; n0 += 64) {
    if (n0 + 64 < NSEQ) stage(n0 + 64, buf ^ 1);
    const __bf16* ka = Ks[buf] + (wm * 16) * LDT;
    const __bf16* vb = Vt[buf] + (wn * 16) * LDT;
    acc = wmma_bf16(load_frag_bf16(ka, LDT, lane),
                    load_frag_bf16(vb, LDT, lane), acc);
    acc = wmma_bf16(load_frag_bf16(ka + 32, LDT, lane),
                    load_frag_bf16(vb + 32, LDT, lane), acc);
    __syncthreads();
    buf ^= 1;
  }

  const int half = lane >> 4, nc = lane & 15;
  const float scale = 1.0f / (float)NSEQ;
  __bf16* dst = kvt + (long long)bh * 64 * 64;
#pragma unroll
  for (int r = 0; r < 8; ++r) {
    int dk = wm * 16 + half * 8 + r;  // 0..47
    int dv = wn * 16 + nc;            // 0..63
    dst[dv * 64 + dk] = (__bf16)(acc[r] * scale);  // transposed store
  }
  for (int i = tid; i < 64 * 16; i += 384)  // zero K-padding cols dk 48..63
    dst[(i >> 4) * 64 + 48 + (i & 15)] = (__bf16)0.0f;
}

// ---------------- out = (q @ kv) * z per (b,h); z from kmean row ---------
// Block: 64 tokens x 64 cols, 128 threads = 4 waves (2x2 of 32x32).
// A frags straight from global fp32 q (K padded to 64: q's cols 48..63 are
// in-bounds garbage that multiplies kvt's zeroed dk columns -> contributes 0).
// B frags straight from global bf16 kvt (transposed layout = A-frag pattern).
__global__ void __launch_bounds__(128) attn_out_kernel(
    const float* __restrict__ qk, const __bf16* __restrict__ kvt,
    float* __restrict__ attn) {
  const int bh = blockIdx.y, b = bh >> 3, h = bh & 7;
  const int tid = threadIdx.x, lane = tid & 31, wave = tid >> 5;
  const int wm = wave >> 1, wn = wave & 1;
  __shared__ float res[64 * 64];
  const long long tok0 = (long long)b * NSEQ + (long long)blockIdx.x * 64;
  const float* qbase = qk + tok0 * (2 * CDIM) + h * DHEAD;
  const __bf16* kvb = kvt + (long long)bh * 64 * 64;

  v8f c00 = {}, c01 = {}, c10 = {}, c11 = {};
#pragma unroll
  for (int k0 = 0; k0 < 64; k0 += 32) {
    v16bf a0 = load_frag_f32(qbase + (long long)(wm * 32) * (2 * CDIM) + k0, 2 * CDIM, lane);
    v16bf a1 = load_frag_f32(qbase + (long long)(wm * 32 + 16) * (2 * CDIM) + k0, 2 * CDIM, lane);
    v16bf b0 = load_frag_bf16(kvb + (wn * 32) * 64 + k0, 64, lane);
    v16bf b1 = load_frag_bf16(kvb + (wn * 32 + 16) * 64 + k0, 64, lane);
    c00 = wmma_bf16(a0, b0, c00);
    c01 = wmma_bf16(a0, b1, c01);
    c10 = wmma_bf16(a1, b0, c10);
    c11 = wmma_bf16(a1, b1, c11);
  }
  const int half = lane >> 4, nc = lane & 15;
  auto st = [&](v8f c, int mb, int nb) {
#pragma unroll
    for (int r = 0; r < 8; ++r)
      res[(wm * 32 + mb + half * 8 + r) * 64 + wn * 32 + nb + nc] = c[r];
  };
  st(c00, 0, 0);  st(c01, 0, 16);
  st(c10, 16, 0); st(c11, 16, 16);
  __syncthreads();

  for (int i = tid; i < 64 * 48; i += 128) {
    int t = i / 48, dv = i - t * 48;
    float z = 1.0f / (res[t * 64 + 48] + 1e-6f);  // q . kmean
    attn[(tok0 + t) * CDIM + h * DHEAD + dv] = res[t * 64 + dv] * z;
  }
}

// ---------------- driver --------------------------------------------------
extern "C" void kernel_launch(void* const* d_in, const int* in_sizes, int n_in,
                              void* d_out, int out_size, void* d_ws, size_t ws_size,
                              hipStream_t stream) {
  const float* x_in    = (const float*)d_in[0];
  const float* cpe1_w  = (const float*)d_in[3];
  const float* cpe1_b  = (const float*)d_in[4];
  const float* norm1_g = (const float*)d_in[5];
  const float* norm1_b = (const float*)d_in[6];
  const float* inp_w   = (const float*)d_in[7];
  const float* inp_b   = (const float*)d_in[8];
  const float* actp_w  = (const float*)d_in[9];
  const float* actp_b  = (const float*)d_in[10];
  const float* dwc_w   = (const float*)d_in[11];
  const float* dwc_b   = (const float*)d_in[12];
  const float* qk_w    = (const float*)d_in[13];
  const float* qk_b    = (const float*)d_in[14];
  const float* lepe_w  = (const float*)d_in[15];
  const float* lepe_b  = (const float*)d_in[16];
  const float* outp_w  = (const float*)d_in[17];
  const float* outp_b  = (const float*)d_in[18];
  const float* cpe2_w  = (const float*)d_in[19];
  const float* cpe2_b  = (const float*)d_in[20];
  const float* norm2_g = (const float*)d_in[21];
  const float* norm2_b = (const float*)d_in[22];
  const float* fc1_w   = (const float*)d_in[23];
  const float* fc1_b   = (const float*)d_in[24];
  const float* fc2_w   = (const float*)d_in[25];
  const float* fc2_b   = (const float*)d_in[26];
  float* out = (float*)d_out;

  // workspace layout (fp32 slots, reused aggressively)
  float* ws    = (float*)d_ws;
  float* x1    = ws;            // shortcut / later x2
  float* xn    = ws + 1 * TC;   // LN1 out / later attn out
  float* act   = ws + 2 * TC;   // silu(act_proj)
  float* hp    = ws + 3 * TC;   // in_proj out / later gated g / later LN2 out
  float* hs    = ws + 4 * TC;   // silu(dwc) attn input / later x3
  float* qkb   = ws + 5 * TC;   // (B,N,2C): qk / later fc1 out
  __bf16* kvt  = (__bf16*)(ws + 7 * TC);  // 128 x 64 x 64 bf16 (1 MB), transposed

  const int M = (int)TTOK;                                // 65536 rows
  const int convBlocks = (int)(TTOK * (CDIM / 4) / 256);  // 24576
  dim3 g384(CDIM / 64, M / 128);
  dim3 g768(2 * CDIM / 64, M / 128);

  // 1) CPE1 residual
  dwconv_kernel<0><<<convBlocks, 256, 0, stream>>>(x_in, cpe1_w, cpe1_b, nullptr, nullptr, x1);
  // 2) LN1
  ln_kernel<<<M / 8, 256, 0, stream>>>(x1, norm1_g, norm1_b, xn);
  // 3) act_res = silu(xn @ act_proj)
  gemm_kernel<ACT_SILU, false><<<g384, 256, 0, stream>>>(xn, actp_w, actp_b, nullptr, act, M, CDIM, CDIM);
  // 4) hp = xn @ in_proj
  gemm_kernel<ACT_NONE, false><<<g384, 256, 0, stream>>>(xn, inp_w, inp_b, nullptr, hp, M, CDIM, CDIM);
  // 5) hs = silu(dwconv(hp))
  dwconv_kernel<1><<<convBlocks, 256, 0, stream>>>(hp, dwc_w, dwc_b, nullptr, nullptr, hs);
  // 6) qk = elu(hs @ qk_w)+1   (2C wide)
  gemm_kernel<ACT_ELU1, false><<<g768, 256, 0, stream>>>(hs, qk_w, qk_b, nullptr, qkb, M, CDIM, 2 * CDIM);
  // 7) kv (+kmean via ones column) per (b,h), transposed bf16 out
  kv_kernel<<<BDIM * NHEAD, 384, 0, stream>>>(qkb, hs, kvt);
  // 8) attn out = (q @ kv) * z  -> xn
  attn_out_kernel<<<dim3(NSEQ / 64, BDIM * NHEAD), 128, 0, stream>>>(qkb, kvt, xn);
  // 9) g = (attn + lepe(hs)) * act_res -> hp
  dwconv_kernel<2><<<convBlocks, 256, 0, stream>>>(hs, lepe_w, lepe_b, xn, act, hp);
  // 10) x2 = shortcut + g @ out_proj (in-place over x1: resid read in epilogue only)
  gemm_kernel<ACT_NONE, true><<<g384, 256, 0, stream>>>(hp, outp_w, outp_b, x1, x1, M, CDIM, CDIM);
  // 11) x3 = x2 + cpe2(x2) -> hs
  dwconv_kernel<0><<<convBlocks, 256, 0, stream>>>(x1, cpe2_w, cpe2_b, nullptr, nullptr, hs);
  // 12) LN2 -> hp
  ln_kernel<<<M / 8, 256, 0, stream>>>(hs, norm2_g, norm2_b, hp);
  // 13) m1 = gelu(m @ fc1) -> qkb (768 wide)
  gemm_kernel<ACT_GELU, false><<<g768, 256, 0, stream>>>(hp, fc1_w, fc1_b, nullptr, qkb, M, CDIM, 2 * CDIM);
  // 14) out = x3 + m1 @ fc2
  gemm_kernel<ACT_NONE, true><<<g384, 256, 0, stream>>>(qkb, fc2_w, fc2_b, hs, out, M, 2 * CDIM, CDIM);
}
